// GIN_20804821582443
// MI455X (gfx1250) — matrix operand order repaired and checked by reading, not compiled, not run on hardware
//
#include <hip/hip_runtime.h>
#include <hip/hip_bf16.h>

// ---------------- problem constants (match reference) ----------------
constexpr int N_NODES = 100000;
constexpr int N_EDGES = 1600000;
constexpr int N_GRAPH = 512;
constexpr int F_IN    = 64;
constexpr int DH      = 128;
constexpr int DHL     = 256;
constexpr int NC      = 128;
constexpr int FP      = 6144;
constexpr int RD      = DH + DH + DHL + FP;   // 6656
constexpr int RDH     = RD / 2;               // 3328
constexpr float BN_EPS = 1e-5f;

// ---------------- WMMA types ----------------
typedef __attribute__((ext_vector_type(16))) __bf16 v16bf;
typedef __attribute__((ext_vector_type(8)))  float  v8f;

// float -> bf16 via native compiler conversion (fptrunc float->bfloat, RNE).
// gfx1250 has native bf16 datapaths; this should select a hardware convert
// instead of the 4-op software rounding sequence.
__device__ __forceinline__ __bf16 f2bf(float f) {
    return (__bf16)f;
}

__device__ __forceinline__ float relu(float v) { return fmaxf(v, 0.0f); }

// A-fragment swizzled element offset for (m, k) inside a 16 x K tile (K % 32 == 0).
// Per ISA 7.12.2 (16-bit A 16x32): lane = m + 16*(k%32 has bit 3 set);
// within-lane slot j = (k&7) + 8*((k&16)!=0); fragment kb = k/32 occupies 512 elems.
__device__ __forceinline__ int a_swz(int m, int k) {
    int kb = k >> 5;
    int kk = k & 31;
    int ln = m + ((kk & 8) ? 16 : 0);
    int j  = (kk & 7) + ((kk & 16) ? 8 : 0);
    return kb * 512 + ln * 16 + j;
}

__device__ __forceinline__ v8f wmma_bf16(v16bf a, v16bf b, v8f c) {
    return __builtin_amdgcn_wmma_f32_16x16x32_bf16(
        false, a, false, b, (short)0, c, false, false);
}

// ---------------- one-shot conversion / swizzle kernels ----------------
// Weights W [K x NOUT] f32 row-major -> bf16 in B-fragment order:
// frag f = ct*(K/32)+kb ; lane l holds column n = ct*16 + (l&15), kh = (l&16)?8:0,
// slots 0..7 -> k = kb*32+kh+j ; slots 8..15 -> k = kb*32+16+kh+(j-8).
__global__ void wswz_kernel(const float* __restrict__ W, __bf16* __restrict__ WB,
                            int K, int NOUT) {
    int t = blockIdx.x * blockDim.x + threadIdx.x;     // one thread per (frag, lane)
    int KB = K >> 5;
    int total = (NOUT >> 4) * KB * 32;
    if (t >= total) return;
    int lane = t & 31;
    int kb   = (t >> 5) % KB;
    int ct   = (t >> 5) / KB;
    int n  = ct * 16 + (lane & 15);
    int ks = kb * 32 + ((lane & 16) ? 8 : 0);
    __bf16* dst = WB + (size_t)t * 16;
#pragma unroll
    for (int j = 0; j < 16; ++j)
        dst[j] = f2bf(W[(size_t)(ks + j + ((j & 8) ? 8 : 0)) * NOUT + n]);
}

// Activations A [M x K] f32 row-major -> bf16 in A-fragment order:
// frag f = rt*(K/32)+kb ; lane l: m = rt*16 + (l&15), kh = (l&16)?8:0,
// slots 0..7 -> k = kb*32+kh+j ; slots 8..15 -> k = kb*32+16+kh+(j-8).
__global__ void aswz_kernel(const float* __restrict__ A, __bf16* __restrict__ AB,
                            int M, int K) {
    int t = blockIdx.x * blockDim.x + threadIdx.x;     // one thread per (frag, lane)
    int KB = K >> 5;
    int total = (M >> 4) * KB * 32;
    if (t >= total) return;
    int lane = t & 31;
    int kb   = (t >> 5) % KB;
    int rt   = (t >> 5) / KB;
    int m  = rt * 16 + (lane & 15);
    int kh = (lane & 16) ? 8 : 0;
    int k0 = kb * 32 + kh;
    const float* srcp = A + (size_t)m * K;
    __bf16* dst = AB + (size_t)t * 16;
#pragma unroll
    for (int j = 0; j < 8; ++j) {
        dst[j]     = f2bf(srcp[k0 + j]);
        dst[8 + j] = f2bf(srcp[k0 + 16 + j]);
    }
}

// ---------------- utility kernels ----------------
__global__ void zero_kernel(float* __restrict__ p, long n) {
    long t = (long)blockIdx.x * blockDim.x + threadIdx.x;
    if (t < n) p[t] = 0.0f;
}

// Edge-parallel scatter-add: aggr[dst] += x[src], feature width F, float4 chunks.
template<int F>
__global__ void scatter_add_kernel(const int* __restrict__ src, const int* __restrict__ dst,
                                   const float* __restrict__ x, float* __restrict__ aggr,
                                   long nWork) {
    long t = (long)blockIdx.x * blockDim.x + threadIdx.x;
    constexpr int CH = F / 4;
    if (t >= nWork) return;
    long e = t / CH;
    int  c = (int)(t % CH) * 4;
    int  s = src[e];
    int  d = dst[e];
    const float4 v = *(const float4*)(x + (long)s * F + c);
    float* p = aggr + (long)d * F + c;
    atomicAdd(p + 0, v.x);
    atomicAdd(p + 1, v.y);
    atomicAdd(p + 2, v.z);
    atomicAdd(p + 3, v.w);
}

// ---------------- fused GIN MLP: relu(bn( relu((x+aggr)@W1+b1) @ W2 + b2 )) ----------------
// One block = 16 nodes, 8 waves. Weights arrive pre-swizzled bf16 (B-fragment order);
// LDS tiles are kept in A-fragment order so every WMMA operand is a contiguous 32B/lane load.
template<int K1, int HID, int OUT>
__global__ __launch_bounds__(256)
void gin_mlp_kernel(const float* __restrict__ xin, const float* __restrict__ aggr,
                    const __bf16* __restrict__ w1b, const float* __restrict__ b1,
                    const __bf16* __restrict__ w2b, const float* __restrict__ b2,
                    const float* __restrict__ bng, const float* __restrict__ bnb,
                    float* __restrict__ hout, int nNodes) {
    __shared__ __align__(32) __bf16 sA[16 * K1];
    __shared__ __align__(32) __bf16 sH[16 * HID];
    const int tid   = threadIdx.x;
    const int lane  = tid & 31;
    const int wave  = tid >> 5;
    const int node0 = blockIdx.x * 16;

    // load (x + aggr) 16xK1 tile -> LDS (bf16, A-fragment swizzled); float4 granularity
    // (4 consecutive k always land in 4 consecutive swizzled slots)
    for (int idx = tid; idx < 16 * (K1 / 4); idx += 256) {
        int r = idx / (K1 / 4);
        int c = (idx % (K1 / 4)) * 4;
        int node = node0 + r;
        float4 v = make_float4(0.f, 0.f, 0.f, 0.f);
        if (node < nNodes) {
            float4 a4 = *(const float4*)(xin  + (size_t)node * K1 + c);
            float4 g4 = *(const float4*)(aggr + (size_t)node * K1 + c);
            v = make_float4(a4.x + g4.x, a4.y + g4.y, a4.z + g4.z, a4.w + g4.w);
        }
        __bf16* d = sA + a_swz(r, c);
        d[0] = f2bf(v.x); d[1] = f2bf(v.y); d[2] = f2bf(v.z); d[3] = f2bf(v.w);
    }
    __syncthreads();

    constexpr int KB1 = K1 / 32;
    constexpr int KB2 = HID / 32;
    const v16bf* SA = (const v16bf*)sA;
    const v16bf* W1 = (const v16bf*)w1b;

    // stage 1: h = relu(A @ W1 + b1) -> LDS (A-fragment swizzled for stage 2)
    for (int ct = wave; ct < HID / 16; ct += 8) {
        v8f acc = {};
#pragma unroll
        for (int kb = 0; kb < KB1; ++kb) {
            v16bf a = SA[kb * 32 + lane];
            v16bf b = W1[(ct * KB1 + kb) * 32 + lane];
            acc = wmma_bf16(a, b, acc);
        }
        const int n = ct * 16 + (lane & 15);
        const float bias = b1[n];
        // swizzled destination components that only depend on n:
        const int kbn = n >> 5;
        const int lno = (n & 8) ? 16 : 0;
        const int j   = ((n & 31) & 7) + ((n & 16) ? 8 : 0);
#pragma unroll
        for (int r = 0; r < 8; ++r) {
            int m = r + ((lane & 16) ? 8 : 0);
            sH[kbn * 512 + (m + lno) * 16 + j] = f2bf(relu(acc[r] + bias));
        }
    }
    __syncthreads();

    const v16bf* SH = (const v16bf*)sH;
    const v16bf* W2 = (const v16bf*)w2b;

    // stage 2: out = relu(bn(h @ W2 + b2))
    for (int ct = wave; ct < OUT / 16; ct += 8) {
        v8f acc = {};
#pragma unroll
        for (int kb = 0; kb < KB2; ++kb) {
            v16bf a = SH[kb * 32 + lane];
            v16bf b = W2[(ct * KB2 + kb) * 32 + lane];
            acc = wmma_bf16(a, b, acc);
        }
        const int n = ct * 16 + (lane & 15);
        const float bias = b2[n];
        const float sc = bng[n] * rsqrtf(1.0f + BN_EPS);
        const float sb = bnb[n];
#pragma unroll
        for (int r = 0; r < 8; ++r) {
            int m = r + ((lane & 16) ? 8 : 0);
            int node = node0 + m;
            if (node < nNodes) {
                hout[(size_t)node * OUT + n] = relu(fmaf(acc[r] + bias, sc, sb));
            }
        }
    }
}

// ---------------- readout pooling ----------------
// hcat layout per graph row: [p1(128) | p2(128) | p3(256) | fingerprint(6144)]
__global__ void pool_kernel(const float* __restrict__ h1, const float* __restrict__ h2,
                            const float* __restrict__ h3, const int* __restrict__ batch,
                            float* __restrict__ hcat) {
    long t = (long)blockIdx.x * blockDim.x + threadIdx.x;
    long total = (long)N_NODES * 512;
    if (t >= total) return;
    int n = (int)(t / 512);
    int c = (int)(t % 512);
    int g = batch[n];
    float v;
    if (c < 128)      v = h1[(long)n * 128 + c];
    else if (c < 256) v = h2[(long)n * 128 + (c - 128)];
    else              v = h3[(long)n * 256 + (c - 256)];
    atomicAdd(&hcat[(long)g * RD + c], v);
}

__global__ void fp_copy_kernel(const float* __restrict__ fp, float* __restrict__ hcat) {
    long t = (long)blockIdx.x * blockDim.x + threadIdx.x;
    if (t >= (long)N_GRAPH * FP) return;
    int g = (int)(t / FP);
    int j = (int)(t % FP);
    hcat[(long)g * RD + 512 + j] = fp[t];
}

// ---------------- dense readout GEMM (WMMA), pre-swizzled bf16 A and B ----------------
// grid.x = M/16 row tiles, grid.y*8 waves = NOUT/16 col tiles
template<int K, int NOUT, bool BN_RELU>
__global__ __launch_bounds__(256)
void dense_kernel(const __bf16* __restrict__ AB, const __bf16* __restrict__ WB,
                  const float* __restrict__ bias, const float* __restrict__ g,
                  const float* __restrict__ bb, float* __restrict__ out) {
    constexpr int KB = K / 32;
    const int lane = threadIdx.x & 31;
    const int wave = threadIdx.x >> 5;
    const int rt   = blockIdx.x;
    const int ct   = blockIdx.y * 8 + wave;
    const v16bf* A = (const v16bf*)AB + (size_t)rt * KB * 32;
    const v16bf* W = (const v16bf*)WB + (size_t)ct * KB * 32;

    v8f acc = {};
#pragma unroll 4
    for (int kb = 0; kb < KB; ++kb) {
        if (kb + 8 < KB)   // speculative prefetch of weight fragments -> global_prefetch_b8
            __builtin_prefetch((const void*)(W + (kb + 8) * 32 + lane), 0, 1);
        v16bf a = A[kb * 32 + lane];
        v16bf b = W[kb * 32 + lane];
        acc = wmma_bf16(a, b, acc);
    }

    const int n = ct * 16 + (lane & 15);
    const float bi = bias[n];
    float sc = 1.0f, sb = 0.0f;
    if (BN_RELU) { sc = g[n] * rsqrtf(1.0f + BN_EPS); sb = bb[n]; }
#pragma unroll
    for (int r = 0; r < 8; ++r) {
        int m = rt * 16 + r + ((lane & 16) ? 8 : 0);
        float v = acc[r] + bi;
        if (BN_RELU) v = relu(fmaf(v, sc, sb));
        out[(size_t)m * NOUT + n] = v;
    }
}

// ---------------- host-side launch ----------------
static inline int cdiv(long a, long b) { return (int)((a + b - 1) / b); }

static inline void launch_wswz(const float* W, __bf16* WB, int K, int NOUT,
                               hipStream_t stream) {
    long total = (long)(NOUT / 16) * (K / 32) * 32;
    wswz_kernel<<<cdiv(total, 256), 256, 0, stream>>>(W, WB, K, NOUT);
}

extern "C" void kernel_launch(void* const* d_in, const int* in_sizes, int n_in,
                              void* d_out, int out_size, void* d_ws, size_t ws_size,
                              hipStream_t stream) {
    // inputs (setup_inputs order)
    const float* x     = (const float*)d_in[0];
    const int*   ei    = (const int*)d_in[1];
    const int*   batch = (const int*)d_in[2];
    const float* fp    = (const float*)d_in[3];
    const float* c1w1  = (const float*)d_in[4];
    const float* c1b1  = (const float*)d_in[5];
    const float* c1w2  = (const float*)d_in[6];
    const float* c1b2  = (const float*)d_in[7];
    const float* c2w1  = (const float*)d_in[8];
    const float* c2b1  = (const float*)d_in[9];
    const float* c2w2  = (const float*)d_in[10];
    const float* c2b2  = (const float*)d_in[11];
    const float* c3w1  = (const float*)d_in[12];
    const float* c3b1  = (const float*)d_in[13];
    const float* c3w2  = (const float*)d_in[14];
    const float* c3b2  = (const float*)d_in[15];
    const float* bn1g  = (const float*)d_in[16];
    const float* bn1b  = (const float*)d_in[17];
    const float* bn2g  = (const float*)d_in[18];
    const float* bn2b  = (const float*)d_in[19];
    const float* bn3g  = (const float*)d_in[20];
    const float* bn3b  = (const float*)d_in[21];
    const float* o1w   = (const float*)d_in[22];
    const float* o1b   = (const float*)d_in[23];
    const float* obng  = (const float*)d_in[24];
    const float* obnb  = (const float*)d_in[25];
    const float* o3w   = (const float*)d_in[26];
    const float* o3b   = (const float*)d_in[27];

    const int* src = ei;             // edge_index[0]
    const int* dst = ei + N_EDGES;   // edge_index[1]

    // ---- workspace partition ----
    // f32 region
    float* ws   = (float*)d_ws;
    float* aggr = ws;                                   // N*128
    float* h1   = aggr + (size_t)N_NODES * 128;         // N*128
    float* h2   = h1   + (size_t)N_NODES * 128;         // N*128
    float* h3   = h2   + (size_t)N_NODES * 128;         // N*256
    float* hcat = h3   + (size_t)N_NODES * 256;         // G*RD
    float* mid  = hcat + (size_t)N_GRAPH * RD;          // G*RDH
    // bf16 region (all sub-buffers are multiples of 512 elements -> 32B alignment holds)
    __bf16* bws   = (__bf16*)(mid + (size_t)N_GRAPH * RDH);
    __bf16* c1w1b = bws;                                // F_IN*DH
    __bf16* c1w2b = c1w1b + (size_t)F_IN * DH;          // DH*DH
    __bf16* c2w1b = c1w2b + (size_t)DH * DH;
    __bf16* c2w2b = c2w1b + (size_t)DH * DH;
    __bf16* c3w1b = c2w2b + (size_t)DH * DH;            // DH*DHL
    __bf16* c3w2b = c3w1b + (size_t)DH * DHL;           // DHL*DHL
    __bf16* o1wb  = c3w2b + (size_t)DHL * DHL;          // RD*RDH
    __bf16* o3wb  = o1wb  + (size_t)RD * RDH;           // RDH*NC
    __bf16* hcatb = o3wb  + (size_t)RDH * NC;           // G*RD
    __bf16* midb  = hcatb + (size_t)N_GRAPH * RD;       // G*RDH

    const int ZB = 256;
    const int nodeBlocks = cdiv(N_NODES, 16);           // 6250

    // ---- one-shot weight convert + swizzle to bf16 B-fragment order ----
    launch_wswz(c1w1, c1w1b, F_IN, DH,  stream);
    launch_wswz(c1w2, c1w2b, DH,  DH,  stream);
    launch_wswz(c2w1, c2w1b, DH,  DH,  stream);
    launch_wswz(c2w2, c2w2b, DH,  DH,  stream);
    launch_wswz(c3w1, c3w1b, DH,  DHL, stream);
    launch_wswz(c3w2, c3w2b, DHL, DHL, stream);
    launch_wswz(o1w,  o1wb,  RD,  RDH, stream);
    launch_wswz(o3w,  o3wb,  RDH, NC,  stream);

    // ---- conv1 (K=64 -> 128 -> 128) ----
    {
        long zn = (long)N_NODES * F_IN;
        zero_kernel<<<cdiv(zn, ZB), ZB, 0, stream>>>(aggr, zn);
        long w = (long)N_EDGES * (F_IN / 4);
        scatter_add_kernel<F_IN><<<cdiv(w, ZB), ZB, 0, stream>>>(src, dst, x, aggr, w);
        gin_mlp_kernel<F_IN, DH, DH><<<nodeBlocks, 256, 0, stream>>>(
            x, aggr, c1w1b, c1b1, c1w2b, c1b2, bn1g, bn1b, h1, N_NODES);
    }
    // ---- conv2 (128 -> 128 -> 128) ----
    {
        long zn = (long)N_NODES * DH;
        zero_kernel<<<cdiv(zn, ZB), ZB, 0, stream>>>(aggr, zn);
        long w = (long)N_EDGES * (DH / 4);
        scatter_add_kernel<DH><<<cdiv(w, ZB), ZB, 0, stream>>>(src, dst, h1, aggr, w);
        gin_mlp_kernel<DH, DH, DH><<<nodeBlocks, 256, 0, stream>>>(
            h1, aggr, c2w1b, c2b1, c2w2b, c2b2, bn2g, bn2b, h2, N_NODES);
    }
    // ---- conv3 (128 -> 256 -> 256) ----
    {
        long zn = (long)N_NODES * DH;
        zero_kernel<<<cdiv(zn, ZB), ZB, 0, stream>>>(aggr, zn);
        long w = (long)N_EDGES * (DH / 4);
        scatter_add_kernel<DH><<<cdiv(w, ZB), ZB, 0, stream>>>(src, dst, h2, aggr, w);
        gin_mlp_kernel<DH, DHL, DHL><<<nodeBlocks, 256, 0, stream>>>(
            h2, aggr, c3w1b, c3b1, c3w2b, c3b2, bn3g, bn3b, h3, N_NODES);
    }
    // ---- readout: pool + concat ----
    {
        long zn = (long)N_GRAPH * RD;
        zero_kernel<<<cdiv(zn, ZB), ZB, 0, stream>>>(hcat, zn);
        long fpw = (long)N_GRAPH * FP;
        fp_copy_kernel<<<cdiv(fpw, ZB), ZB, 0, stream>>>(fp, hcat);
        long pw = (long)N_NODES * 512;
        pool_kernel<<<cdiv(pw, ZB), ZB, 0, stream>>>(h1, h2, h3, batch, hcat);
    }
    // ---- dense readout layers (A pre-swizzled to bf16 fragments first) ----
    {
        long an1 = (long)(N_GRAPH / 16) * (RD / 32) * 32;
        aswz_kernel<<<cdiv(an1, ZB), ZB, 0, stream>>>(hcat, hcatb, N_GRAPH, RD);
        dim3 g1(N_GRAPH / 16, RDH / (16 * 8));          // 32 x 26
        dense_kernel<RD, RDH, true><<<g1, 256, 0, stream>>>(
            hcatb, o1wb, o1b, obng, obnb, mid);

        long an2 = (long)(N_GRAPH / 16) * (RDH / 32) * 32;
        aswz_kernel<<<cdiv(an2, ZB), ZB, 0, stream>>>(mid, midb, N_GRAPH, RDH);
        dim3 g2(N_GRAPH / 16, NC / (16 * 8));           // 32 x 1
        dense_kernel<RDH, NC, false><<<g2, 256, 0, stream>>>(
            midb, o3wb, o3b, nullptr, nullptr, (float*)d_out);
    }
}
